// Pointer_network_7919919694056
// MI455X (gfx1250) — matrix operand order repaired and checked by reading, not compiled, not run on hardware
//
#include <hip/hip_runtime.h>

// ---------------------------------------------------------------------------
// Pointer network (B=1024, S=H=256, I=1) for MI455X (gfx1250, wave32, WMMA).
// - All GEMMs use v_wmma_f32_16x16x32_bf16, 16x32 output tile per wave
//   (two accumulators share one A fragment -> fewer loads per WMMA).
// - We[B,S,H] kept in bf16 (128MB) so the 256 decoder attention sweeps run
//   out of the 192MB L2 instead of HBM.
// - tanh via hardware v_exp_f32 identity: tanh(x) = 1 - 2/(exp(2x)+1).
// ---------------------------------------------------------------------------

typedef __bf16 bf16;
typedef __attribute__((ext_vector_type(16))) __bf16 v16bf;
typedef __attribute__((ext_vector_type(8)))  __bf16 v8bf;
typedef __attribute__((ext_vector_type(8)))  float  v8f;

__device__ __forceinline__ float sigmoidf_(float x) { return 1.0f / (1.0f + __expf(-x)); }
__device__ __forceinline__ float tanhf_(float x)    { return 1.0f - 2.0f / (__expf(2.0f * x) + 1.0f); }

// ---------------------------------------------------------------------------
// GEMM: out[M,N](f32) = A[M,K](bf16) * W[N,K]^T(bf16) + bias[N]
//       (+ optional rank-1 term xvec[m]*wihcol[n], since input dim I==1)
// One wave computes a 16x32 tile (two 16x16 WMMA accumulators, shared A).
// Grid must cover exactly (M/16)*(N/32) waves (4 waves / 128-thread block).
// ---------------------------------------------------------------------------
__global__ void k_gemm_gates(const bf16* __restrict__ A, const bf16* __restrict__ W,
                             const float* __restrict__ bias,
                             const float* __restrict__ xvec, int xstride, int xoff,
                             const float* __restrict__ wihcol,
                             float* __restrict__ out, int N, int K)
{
    const int wave   = blockIdx.x * (blockDim.x >> 5) + (threadIdx.x >> 5);
    const int tilesN = N >> 5;                 // 32-wide column groups
    const int tm     = (wave / tilesN) << 4;
    const int tn     = (wave % tilesN) << 5;
    const int lane   = threadIdx.x & 31;
    const int hf     = lane >> 4;
    const int r      = lane & 15;

    const bf16* arow  = A + (size_t)(tm + r) * K;
    const bf16* wrow0 = W + (size_t)(tn + r) * K;
    const bf16* wrow1 = W + (size_t)(tn + 16 + r) * K;

    v8f acc0 = {}, acc1 = {};
    for (int kk = 0; kk < K; kk += 32) {
        union { v16bf v; v8bf h[2]; } ua;
        ua.h[0] = *(const v8bf*)(arow + kk + hf * 16);       // K = kk + hf*16 .. +7
        ua.h[1] = *(const v8bf*)(arow + kk + hf * 16 + 8);   // K = kk+hf*16+8 .. +15
        v16bf b0 = *(const v16bf*)(wrow0 + kk + hf * 16);    // K = kk + hf*16 .. +15
        v16bf b1 = *(const v16bf*)(wrow1 + kk + hf * 16);
        acc0 = __builtin_amdgcn_wmma_f32_16x16x32_bf16(
            false, ua.v, false, b0, (short)0, acc0, false, false);
        acc1 = __builtin_amdgcn_wmma_f32_16x16x32_bf16(
            false, ua.v, false, b1, (short)0, acc1, false, false);
    }

    const int   n0  = tn + r;
    const int   n1  = tn + 16 + r;
    const float bn0 = bias[n0];
    const float bn1 = bias[n1];
    const float wn0 = (wihcol != nullptr) ? wihcol[n0] : 0.0f;
    const float wn1 = (wihcol != nullptr) ? wihcol[n1] : 0.0f;
#pragma unroll
    for (int j = 0; j < 8; ++j) {
        const int m = tm + j + hf * 8;
        float v0 = acc0[j] + bn0;
        float v1 = acc1[j] + bn1;
        if (xvec != nullptr) {
            const float xm = xvec[(size_t)m * xstride + xoff];
            v0 += xm * wn0;
            v1 += xm * wn1;
        }
        out[(size_t)m * N + n0] = v0;
        out[(size_t)m * N + n1] = v1;
    }
}

// Same GEMM core, bf16 output with row stride (builds We[:, t, :] in-place).
__global__ void k_gemm_bf16out(const bf16* __restrict__ A, const bf16* __restrict__ W,
                               const float* __restrict__ bias,
                               bf16* __restrict__ out, size_t ostride, int N, int K)
{
    const int wave   = blockIdx.x * (blockDim.x >> 5) + (threadIdx.x >> 5);
    const int tilesN = N >> 5;
    const int tm     = (wave / tilesN) << 4;
    const int tn     = (wave % tilesN) << 5;
    const int lane   = threadIdx.x & 31;
    const int hf     = lane >> 4;
    const int r      = lane & 15;

    const bf16* arow  = A + (size_t)(tm + r) * K;
    const bf16* wrow0 = W + (size_t)(tn + r) * K;
    const bf16* wrow1 = W + (size_t)(tn + 16 + r) * K;

    v8f acc0 = {}, acc1 = {};
    for (int kk = 0; kk < K; kk += 32) {
        union { v16bf v; v8bf h[2]; } ua;
        ua.h[0] = *(const v8bf*)(arow + kk + hf * 16);
        ua.h[1] = *(const v8bf*)(arow + kk + hf * 16 + 8);
        v16bf b0 = *(const v16bf*)(wrow0 + kk + hf * 16);
        v16bf b1 = *(const v16bf*)(wrow1 + kk + hf * 16);
        acc0 = __builtin_amdgcn_wmma_f32_16x16x32_bf16(
            false, ua.v, false, b0, (short)0, acc0, false, false);
        acc1 = __builtin_amdgcn_wmma_f32_16x16x32_bf16(
            false, ua.v, false, b1, (short)0, acc1, false, false);
    }

    const int   n0  = tn + r;
    const int   n1  = tn + 16 + r;
    const float bn0 = bias[n0];
    const float bn1 = bias[n1];
#pragma unroll
    for (int j = 0; j < 8; ++j) {
        const int m = tm + j + hf * 8;
        out[(size_t)m * ostride + n0] = (bf16)(acc0[j] + bn0);
        out[(size_t)m * ostride + n1] = (bf16)(acc1[j] + bn1);
    }
}

// LSTM pointwise: gates[B,4H] (i,f,g,o) -> c, h(bf16). One thread per (b,n).
__global__ void k_lstm_pointwise(const float* __restrict__ gates,
                                 float* __restrict__ c, bf16* __restrict__ hbf, int H)
{
    const int idx = blockIdx.x * blockDim.x + threadIdx.x;
    const int b = idx / H, n = idx % H;
    const float* g = gates + (size_t)b * 4 * H;
    const float gi = g[n], gf = g[H + n], gg = g[2 * H + n], go = g[3 * H + n];
    const float cn = sigmoidf_(gf) * c[idx] + sigmoidf_(gi) * tanhf_(gg);
    const float hn = sigmoidf_(go) * tanhf_(cn);
    c[idx]   = cn;
    hbf[idx] = (bf16)hn;
}

// Attention scores: u[b,s] = sum_n tanh(We[b,s,n] + Wd[b,n]) * v[n] + vb.
// One wave per (b,s); We row is 256 bf16 = 512B contiguous, L2-friendly.
__global__ void k_attn(const bf16* __restrict__ We, const float* __restrict__ wd,
                       const float* __restrict__ vw, const float* __restrict__ vb,
                       float* __restrict__ u, int S, int H)
{
    const int wid  = blockIdx.x * (blockDim.x >> 5) + (threadIdx.x >> 5);
    const int lane = threadIdx.x & 31;
    const int b = wid / S, s = wid % S;
    const bf16*  we  = We + ((size_t)b * S + s) * (size_t)H;
    const float* wdb = wd + (size_t)b * H;
    float acc = 0.0f;
    for (int k = lane; k < H; k += 32)
        acc += tanhf_((float)we[k] + wdb[k]) * vw[k];
#pragma unroll
    for (int off = 16; off; off >>= 1) acc += __shfl_xor(acc, off, 32);
    if (lane == 0) u[wid] = acc + vb[0];
}

// Per-row softmax over S, write out[b,t,:], masked hard argmax, mask update,
// gather decoder input. One block (S=256 threads) per batch row.
__global__ void k_softmax_ptr(const float* __restrict__ u, const float* __restrict__ x,
                              float* __restrict__ mask, float* __restrict__ dec_in,
                              float* __restrict__ out, int t, int S)
{
    const int b = blockIdx.x, s = threadIdx.x;
    __shared__ float sf[256];
    __shared__ int   si[256];

    const float v = u[(size_t)b * S + s];
    sf[s] = v; __syncthreads();
    for (int off = S >> 1; off; off >>= 1) {
        if (s < off) sf[s] = fmaxf(sf[s], sf[s + off]);
        __syncthreads();
    }
    const float mx = sf[0]; __syncthreads();

    const float e = __expf(v - mx);
    sf[s] = e; __syncthreads();
    for (int off = S >> 1; off; off >>= 1) {
        if (s < off) sf[s] += sf[s + off];
        __syncthreads();
    }
    const float a = e / sf[0];
    out[(size_t)b * S * S + (size_t)t * S + s] = a;

    const float am = a * mask[(size_t)b * S + s];
    __syncthreads();
    sf[s] = am; si[s] = s; __syncthreads();
    for (int off = S >> 1; off; off >>= 1) {
        if (s < off) {
            const float v2 = sf[s + off]; const int i2 = si[s + off];
            if (v2 > sf[s] || (v2 == sf[s] && i2 < si[s])) { sf[s] = v2; si[s] = i2; }
        }
        __syncthreads();
    }
    if (s == 0) {
        const int idx = si[0];
        mask[(size_t)b * S + idx] = 0.0f;
        dec_in[b] = x[(size_t)b * S + idx];
    }
}

// Per-call init: bf16 weight copies, fused biases, zero state, mask=1.
__global__ void k_prep(const float* encWhh, const float* decWhh,
                       const float* W1w, const float* W2w,
                       const float* enc_bih, const float* enc_bhh,
                       const float* dec_bih, const float* dec_bhh,
                       bf16* encWhh_bf, bf16* decWhh_bf, bf16* W1w_bf, bf16* W2w_bf,
                       float* enc_bias, float* dec_bias,
                       float* c, bf16* hbf, float* mask, int H, int B, int S)
{
    const int tid = blockIdx.x * blockDim.x + threadIdx.x;
    const int nt  = gridDim.x * blockDim.x;
    const int n4hh = 4 * H * H;
    for (int i = tid; i < n4hh; i += nt) {
        encWhh_bf[i] = (bf16)encWhh[i];
        decWhh_bf[i] = (bf16)decWhh[i];
    }
    for (int i = tid; i < H * H; i += nt) {
        W1w_bf[i] = (bf16)W1w[i];
        W2w_bf[i] = (bf16)W2w[i];
    }
    for (int i = tid; i < 4 * H; i += nt) {
        enc_bias[i] = enc_bih[i] + enc_bhh[i];
        dec_bias[i] = dec_bih[i] + dec_bhh[i];
    }
    for (int i = tid; i < B * H; i += nt) { c[i] = 0.0f; hbf[i] = (bf16)0.0f; }
    for (size_t i = tid; i < (size_t)B * S; i += nt) mask[i] = 1.0f;
}

extern "C" void kernel_launch(void* const* d_in, const int* in_sizes, int n_in,
                              void* d_out, int out_size, void* d_ws, size_t ws_size,
                              hipStream_t stream)
{
    (void)n_in; (void)out_size; (void)ws_size;

    const int H  = in_sizes[10];          // W1b -> hidden dim
    const int S  = H;                     // model requires seq_len == hidden_dim
    const int B  = in_sizes[0] / S;
    const int G4 = 4 * H;

    const float* x       = (const float*)d_in[0];
    const float* enc_Wih = (const float*)d_in[1];   // [4H,1] -> column vector
    const float* enc_Whh = (const float*)d_in[2];
    const float* enc_bih = (const float*)d_in[3];
    const float* enc_bhh = (const float*)d_in[4];
    const float* dec_Wih = (const float*)d_in[5];
    const float* dec_Whh = (const float*)d_in[6];
    const float* dec_bih = (const float*)d_in[7];
    const float* dec_bhh = (const float*)d_in[8];
    const float* W1w     = (const float*)d_in[9];
    const float* W1b     = (const float*)d_in[10];
    const float* W2w     = (const float*)d_in[11];
    const float* W2b     = (const float*)d_in[12];
    const float* vw      = (const float*)d_in[13];
    const float* vb      = (const float*)d_in[14];
    float* out = (float*)d_out;

    // ---- workspace carve (256B aligned) ----
    char* p = (char*)d_ws;
    auto carve = [&](size_t bytes) -> char* {
        char* q = p;
        p += (bytes + 255) & ~(size_t)255;
        return q;
    };
    bf16*  encWhh_bf = (bf16*)carve((size_t)G4 * H * 2);
    bf16*  decWhh_bf = (bf16*)carve((size_t)G4 * H * 2);
    bf16*  W1w_bf    = (bf16*)carve((size_t)H * H * 2);
    bf16*  W2w_bf    = (bf16*)carve((size_t)H * H * 2);
    float* enc_bias  = (float*)carve((size_t)G4 * 4);
    float* dec_bias  = (float*)carve((size_t)G4 * 4);
    float* cbuf      = (float*)carve((size_t)B * H * 4);
    bf16*  hbf       = (bf16*)carve((size_t)B * H * 2);
    float* gates     = (float*)carve((size_t)B * G4 * 4);
    bf16*  We        = (bf16*)carve((size_t)B * S * H * 2);   // 128MB, ~L2 resident
    float* wd        = (float*)carve((size_t)B * H * 4);
    float* u         = (float*)carve((size_t)B * S * 4);
    float* mask      = (float*)carve((size_t)B * S * 4);
    float* dec_in    = (float*)carve((size_t)B * 4);

    k_prep<<<512, 256, 0, stream>>>(enc_Whh, dec_Whh, W1w, W2w,
                                    enc_bih, enc_bhh, dec_bih, dec_bhh,
                                    encWhh_bf, decWhh_bf, W1w_bf, W2w_bf,
                                    enc_bias, dec_bias, cbuf, hbf, mask, H, B, S);

    const dim3 blk(128);                                   // 4 waves / block
    const int blocksGates = ((B / 16) * (G4 / 32)) / 4;    // [B,4H] 16x32 tiles
    const int blocksH     = ((B / 16) * (H  / 32)) / 4;    // [B,H]  16x32 tiles
    const int blocksPW    = (B * H) / 256;
    const int attnBlocks  = (B * S) / 8;                   // 8 waves / block

    // ---- encoder: 256 sequential LSTM steps + We[:,t,:] projection ----
    for (int t = 0; t < S; ++t) {
        k_gemm_gates<<<blocksGates, blk, 0, stream>>>(
            hbf, encWhh_bf, enc_bias, x, S, t, enc_Wih, gates, G4, H);
        k_lstm_pointwise<<<blocksPW, 256, 0, stream>>>(gates, cbuf, hbf, H);
        k_gemm_bf16out<<<blocksH, blk, 0, stream>>>(
            hbf, W1w_bf, W1b, We + (size_t)t * H, (size_t)S * H, H, H);
    }

    // ---- decoder init step (zero input) ----
    k_gemm_gates<<<blocksGates, blk, 0, stream>>>(
        hbf, decWhh_bf, dec_bias, nullptr, 0, 0, nullptr, gates, G4, H);
    k_lstm_pointwise<<<blocksPW, 256, 0, stream>>>(gates, cbuf, hbf, H);

    // ---- decoder: 256 pointer steps ----
    for (int t = 0; t < S; ++t) {
        k_gemm_gates<<<blocksH, blk, 0, stream>>>(
            hbf, W2w_bf, W2b, nullptr, 0, 0, nullptr, wd, H, H);
        k_attn<<<attnBlocks, 256, 0, stream>>>(We, wd, vw, vb, u, S, H);
        k_softmax_ptr<<<B, S, 0, stream>>>(u, x, mask, dec_in, out, t, S);
        k_gemm_gates<<<blocksGates, blk, 0, stream>>>(
            hbf, decWhh_bf, dec_bias, dec_in, 1, 0, dec_Wih, gates, G4, H);
        k_lstm_pointwise<<<blocksPW, 256, 0, stream>>>(gates, cbuf, hbf, H);
    }
}
